// RTDETRHead_53609781788851
// MI455X (gfx1250) — matrix operand order repaired and checked by reading, not compiled, not run on hardware
//
#include <hip/hip_runtime.h>
#include <hip/hip_bf16.h>
#include <math.h>

typedef __bf16 v16bf __attribute__((ext_vector_type(16)));
typedef float  v8f   __attribute__((ext_vector_type(8)));
typedef int    v4i   __attribute__((ext_vector_type(4)));
typedef unsigned int u32x4 __attribute__((ext_vector_type(4)));
typedef int          i32x8 __attribute__((ext_vector_type(8)));
typedef int          i32x4 __attribute__((ext_vector_type(4)));

#define AS1 __attribute__((address_space(1)))
#define AS3 __attribute__((address_space(3)))

// ---- feature detection (device pass only; host pass sees fallbacks) -------
#if defined(__has_builtin)
#  if __has_builtin(__builtin_amdgcn_global_load_async_to_lds_b128)
#    define HAVE_ASYNC_LDS 1
#  endif
#  if __has_builtin(__builtin_amdgcn_tensor_load_to_lds) && !__has_include(<hip/amd_detail/amd_gfx1250_TDM.h>)
#    define HAVE_TDM5 1   /* ROCm 7.2 5-arg tensor_load_to_lds */
#  endif
#endif
#ifndef HAVE_ASYNC_LDS
#define HAVE_ASYNC_LDS 0
#endif
#ifndef HAVE_TDM5
#define HAVE_TDM5 0
#endif

namespace {

constexpr int kLq    = 300;
constexpr int kC     = 256;
constexpr int kHeads = 8;
constexpr int kHd    = 32;
constexpr int kLev   = 3;
constexpr int kPts   = 4;
constexpr int kLv    = 8400;   // 80*80 + 40*40 + 20*20
constexpr int kK     = 256;    // shared K dim of every GEMM in this model

__device__ __forceinline__ void split_bf16(float x, __bf16& hi, __bf16& lo) {
  hi = (__bf16)x;
  lo = (__bf16)(x - (float)hi);
}

__device__ __forceinline__ void async_copy_b128(const float* g, float* l) {
#if HAVE_ASYNC_LDS
  __builtin_amdgcn_global_load_async_to_lds_b128(
      (AS1 v4i*)(uintptr_t)g,
      (AS3 v4i*)(unsigned int)(uintptr_t)l, 0, 0);
#else
  *(float4*)l = *(const float4*)g;      // sync fallback: vmem load + ds_store
#endif
}

__device__ __forceinline__ void wait_async0() {
#if defined(__has_builtin) && __has_builtin(__builtin_amdgcn_s_wait_asynccnt)
  __builtin_amdgcn_s_wait_asynccnt(0);
#elif defined(__AMDGCN__)
  asm volatile("s_wait_asynccnt 0x0" ::: "memory");
#endif
}

__device__ __forceinline__ void wait_tensor0() {
#if defined(__has_builtin) && __has_builtin(__builtin_amdgcn_s_wait_tensorcnt)
  __builtin_amdgcn_s_wait_tensorcnt(0);
#elif defined(__AMDGCN__)
  asm volatile("s_wait_tensorcnt 0x0" ::: "memory");
#endif
}

// C[M x Ncols] = A[M x K] @ B[K x Ncols] + bias (K = 256).
// Block = 8 wave32 waves = 128(M) x 16(N) output tile.
//  - B panel (256x16 f32, 16KB) staged to LDS once: TDM tensor_load_to_lds
//    when available, else cooperative async b128 copies.
//  - A tile (128x32 f32, 16KB) double-buffered in LDS via async b128 copies,
//    prefetching k-step i+1 while step i feeds the WMMAs.
//  - f32 operands split hi/lo bf16 -> 3x v_wmma_f32_16x16x32_bf16 (near-f32).
template <bool OUT_BF16>
__global__ __launch_bounds__(256)
void gemm_bias_wmma(const float* __restrict__ A,
                    const float* __restrict__ B,
                    const float* __restrict__ bias,
                    void* __restrict__ Cout,
                    int M, int Ncols)
{
  __shared__ alignas(16) float Bsh[kK * 16];        // 16 KB
  __shared__ alignas(16) float Ash[2][128 * 32];    // 2 x 16 KB

  const int lane = threadIdx.x;                // 0..31
  const int wave = threadIdx.y;                // 0..7
  const int tid  = wave * 32 + lane;
  const int tileN = blockIdx.x;
  const int Mtiles = M >> 4;
  int tileM = blockIdx.y * 8 + wave;
  const bool valid = tileM < Mtiles;
  if (!valid) tileM = Mtiles - 1;              // clamp: all waves stay for barriers
  const int blockRow0 = blockIdx.y * 128;

  const int mh   = lane & 15;
  const int half = lane >> 4;

  // ---- stage B panel (K x 16) into LDS ------------------------------------
#if HAVE_TDM5
  if (wave == 0) {
    const unsigned long long ga = (unsigned long long)(uintptr_t)(B + (size_t)tileN * 16);
    const unsigned int lds_off  = (unsigned int)(uintptr_t)Bsh;
    u32x4 g0;
    g0[0] = 1u;                                        // count=1 valid descriptor
    g0[1] = lds_off;                                   // lds_addr (bytes)
    g0[2] = (unsigned int)ga;                          // global_addr[31:0]
    g0[3] = (unsigned int)(ga >> 32) | 0x80000000u;    // global_addr[56:32] | type=2
    const unsigned long long s0 = (unsigned long long)(unsigned)Ncols;
    i32x8 g1;
    g1[0] = (int)(2u << 16);                                       // data_size = 4B
    g1[1] = (int)(((unsigned)Ncols & 0xffffu) << 16);              // tensor_dim0[15:0]
    g1[2] = (int)((((unsigned)Ncols >> 16) & 0xffffu) | ((unsigned)kK << 16)); // dim0 hi | dim1 lo
    g1[3] = (int)(16u << 16);                                      // dim1 hi=0 | tile_dim0=16
    g1[4] = (int)(unsigned)kK;                                     // tile_dim1=256, tile_dim2=0
    g1[5] = (int)(unsigned)(s0 & 0xffffffffu);                     // tensor_dim0_stride lo
    g1[6] = (int)(unsigned)((s0 >> 32) & 0xffffu);                 // stride0 hi | stride1 lo=0
    g1[7] = 0;
    i32x4 z4 = {0, 0, 0, 0};
    __builtin_amdgcn_tensor_load_to_lds(g0, g1, z4, z4, 0);
  }
#else
  {
    // thread t stages B row t (16 contiguous floats = 4 float4)
    const float* gB = B + (size_t)tid * Ncols + (size_t)tileN * 16;
    float* lB = Bsh + tid * 16;
#pragma unroll
    for (int j = 0; j < 4; ++j) async_copy_b128(gB + 4 * j, lB + 4 * j);
  }
#endif

  // ---- A tile staging: 128 rows x 32 K floats, 256 threads ----------------
  auto stage_a = [&](int k0, int bufi) {
    const int rbase = tid >> 3;            // 0..31
    const int cc    = (tid & 7) * 4;       // 0..28 step 4
#pragma unroll
    for (int i = 0; i < 4; ++i) {
      const int r = rbase + 32 * i;        // 0..127
      int grow = blockRow0 + r;
      if (grow >= M) grow = M - 1;         // clamp (duplicate loads, never OOB)
      async_copy_b128(A + (size_t)grow * kK + k0 + cc, &Ash[bufi][r * 32 + cc]);
    }
  };

  stage_a(0, 0);
#if HAVE_TDM5
  if (wave == 0) wait_tensor0();
#endif
  wait_async0();
  __syncthreads();

  v8f acc = {0.f, 0.f, 0.f, 0.f, 0.f, 0.f, 0.f, 0.f};
  int buf = 0;
  for (int k0 = 0; k0 < kK; k0 += 32) {
    if (k0 + 32 < kK) stage_a(k0 + 32, buf ^ 1);   // prefetch next tile (ASYNCcnt)

    // A fragment (16x32 bf16) from LDS: elems 0..7 -> K=half*8+e, 8..15 -> +16
    const float* Arow = &Ash[buf][(wave * 16 + mh) * 32];
    float a[16], b[16];
#pragma unroll
    for (int e = 0; e < 8; ++e) { a[e] = Arow[half * 8 + e]; a[8 + e] = Arow[16 + half * 8 + e]; }
    // B fragment (32x16 bf16) from LDS: elems 0..7 -> K=k0+half*8+e, 8..15 -> +16, col=mh
#pragma unroll
    for (int e = 0; e < 16; ++e) b[e] = Bsh[(k0 + (e >> 3) * 16 + half * 8 + (e & 7)) * 16 + mh];

    v16bf ah, al, bh, bl;
#pragma unroll
    for (int e = 0; e < 16; ++e) {
      __bf16 h, l;
      split_bf16(a[e], h, l); ah[e] = h; al[e] = l;
      split_bf16(b[e], h, l); bh[e] = h; bl[e] = l;
    }
    // (ah+al)(bh+bl) ~= ah*bh + al*bh + ah*bl  (lo*lo term ~2^-16, dropped)
    acc = __builtin_amdgcn_wmma_f32_16x16x32_bf16(false, ah, false, bh, (short)0, acc, false, false);
    acc = __builtin_amdgcn_wmma_f32_16x16x32_bf16(false, al, false, bh, (short)0, acc, false, false);
    acc = __builtin_amdgcn_wmma_f32_16x16x32_bf16(false, ah, false, bl, (short)0, acc, false, false);

    wait_async0();        // prefetched tile landed
    __syncthreads();      // everyone done reading buf before it is overwritten
    buf ^= 1;
  }

  // C/D layout: VGPR r -> row = half*8 + r, col = lane&15
  if (valid) {
    const int   col = tileN * 16 + mh;
    const float bv  = bias ? bias[col] : 0.0f;
#pragma unroll
    for (int r = 0; r < 8; ++r) {
      const int   row = tileM * 16 + half * 8 + r;
      const float v   = acc[r] + bv;
      if (OUT_BF16)
        ((__hip_bfloat16*)Cout)[(size_t)row * Ncols + col] = __float2bfloat16(v);
      else
        ((float*)Cout)[(size_t)row * Ncols + col] = v;
    }
  }
}

// One wave per (n, q, head); lane = channel (hd = 32 == wave32).
// Softmax over the 12 (level,point) logits done redundantly per lane in-reg.
// V is bf16 (N*Lv, 256): 68.8MB -> resident in the 192MB L2; per-corner reads
// are 64B-contiguous across the wave.
__global__ __launch_bounds__(256)
void ms_deform_sample(const float* __restrict__ OFF,   // (N*Lq, 192)
                      const float* __restrict__ ATT,   // (N*Lq, 96) logits
                      const __hip_bfloat16* __restrict__ V,  // (N*Lv, 256)
                      const float* __restrict__ RP,    // (N, Lq, 3, 4)
                      float* __restrict__ outpre)      // (N*Lq, 256)
{
  const int lane = threadIdx.x;
  const int h    = threadIdx.y;
  const int nq   = blockIdx.x;
  const int n    = nq / kLq;

  const float* off = OFF + (size_t)nq * (kHeads * kLev * kPts * 2) + h * (kLev * kPts * 2);
  const float* att = ATT + (size_t)nq * (kHeads * kLev * kPts)     + h * (kLev * kPts);
  const float* rp  = RP  + (size_t)nq * (kLev * 4);

  float logit[kLev * kPts];
  float mx = -1e30f;
#pragma unroll
  for (int i = 0; i < kLev * kPts; ++i) { logit[i] = att[i]; mx = fmaxf(mx, logit[i]); }
  float sum = 0.f;
#pragma unroll
  for (int i = 0; i < kLev * kPts; ++i) { logit[i] = __expf(logit[i] - mx); sum += logit[i]; }
  const float inv = 1.0f / sum;

  const int W_[kLev] = {80, 40, 20};
  const int H_[kLev] = {80, 40, 20};
  const int S_[kLev] = {0, 6400, 8000};

  float accv = 0.f;
  int idx = 0;
#pragma unroll
  for (int lvl = 0; lvl < kLev; ++lvl) {
    const float cx = rp[lvl * 4 + 0], cy = rp[lvl * 4 + 1];
    const float rw = rp[lvl * 4 + 2], rh = rp[lvl * 4 + 3];
    const int W = W_[lvl], Hh = H_[lvl];
    const size_t base = (size_t)n * kLv + S_[lvl];
#pragma unroll
    for (int p = 0; p < kPts; ++p, ++idx) {
      const float dx = off[idx * 2 + 0], dy = off[idx * 2 + 1];
      // loc = ref + off/P * wh * 0.5 ; gx = loc*W - 0.5 (align_corners=False)
      const float lx = cx + dx * 0.125f * rw;
      const float ly = cy + dy * 0.125f * rh;
      const float gx = lx * (float)W  - 0.5f;
      const float gy = ly * (float)Hh - 0.5f;
      const float x0f = floorf(gx), y0f = floorf(gy);
      const int   x0 = (int)x0f, y0 = (int)y0f;
      const float wx = gx - x0f, wy = gy - y0f;
      const float aw = logit[idx] * inv;

      float s = 0.f;
#pragma unroll
      for (int dyc = 0; dyc < 2; ++dyc) {
#pragma unroll
        for (int dxc = 0; dxc < 2; ++dxc) {
          const int ix = x0 + dxc, iy = y0 + dyc;
          const float w2 = (dxc ? wx : 1.f - wx) * (dyc ? wy : 1.f - wy);
          if (ix >= 0 && ix < W && iy >= 0 && iy < Hh) {
            const size_t pos = base + (size_t)iy * W + ix;
            s += w2 * __bfloat162float(V[pos * kC + h * kHd + lane]);
          }
        }
      }
      accv += aw * s;
    }
  }
  outpre[(size_t)nq * kC + h * kHd + lane] = accv;
}

} // namespace

extern "C" void kernel_launch(void* const* d_in, const int* in_sizes, int n_in,
                              void* d_out, int out_size, void* d_ws, size_t ws_size,
                              hipStream_t stream)
{
  const float* query  = (const float*)d_in[0];
  const float* refp   = (const float*)d_in[1];
  const float* value  = (const float*)d_in[2];
  const float* W_off  = (const float*)d_in[3];
  const float* b_off  = (const float*)d_in[4];
  const float* W_attn = (const float*)d_in[5];
  const float* b_attn = (const float*)d_in[6];
  const float* W_val  = (const float*)d_in[7];
  const float* b_val  = (const float*)d_in[8];
  const float* W_out  = (const float*)d_in[9];
  const float* b_out  = (const float*)d_in[10];
  float* out = (float*)d_out;

  const int NQ = 16 * kLq;      // 4800 query rows
  const int MV = 16 * kLv;      // 134400 value rows

  // Workspace layout (~79.3 MB total)
  char* ws = (char*)d_ws;
  float* OFF = (float*)ws;                       ws += (size_t)NQ * 192 * sizeof(float);
  float* ATT = (float*)ws;                       ws += (size_t)NQ * 96  * sizeof(float);
  __hip_bfloat16* Vp = (__hip_bfloat16*)ws;      ws += (size_t)MV * kC  * sizeof(__hip_bfloat16);
  float* OP  = (float*)ws;

  const dim3 blk(32, 8);                          // 8 wave32 waves per block

  // 1) sampling-offset projection: (4800x256)@(256x192)
  gemm_bias_wmma<false><<<dim3(192 / 16, (NQ / 16 + 7) / 8), blk, 0, stream>>>(
      query, W_off, b_off, OFF, NQ, 192);
  // 2) attention-logit projection: (4800x256)@(256x96)
  gemm_bias_wmma<false><<<dim3(96 / 16, (NQ / 16 + 7) / 8), blk, 0, stream>>>(
      query, W_attn, b_attn, ATT, NQ, 96);
  // 3) value projection (dominant GEMM): (134400x256)@(256x256) -> bf16
  gemm_bias_wmma<true><<<dim3(kC / 16, (MV / 16 + 7) / 8), blk, 0, stream>>>(
      value, W_val, b_val, (void*)Vp, MV, kC);
  // 4) softmax + bilinear deformable sampling
  ms_deform_sample<<<dim3(NQ), blk, 0, stream>>>(OFF, ATT, Vp, refp, OP);
  // 5) output projection: (4800x256)@(256x256)
  gemm_bias_wmma<false><<<dim3(kC / 16, (NQ / 16 + 7) / 8), blk, 0, stream>>>(
      OP, W_out, b_out, out, NQ, kC);
}
